// Caps2dMatwo_75582834475478
// MI455X (gfx1250) — matrix operand order repaired and checked by reading, not compiled, not run on hardware
//
#include <hip/hip_runtime.h>
#include <hip/hip_bf16.h>
#include <math.h>

// Fused Caps2dMatwo forward for MI455X (gfx1250, wave32).
// One block = 8 waves = one row-tile of 8 pixels for one (n, h).
// Stage 0: async-stage x halo tile (GLOBAL_LOAD_ASYNC_TO_LDS_B32 when the
//          toolchain exposes it) + normalized routing weights into LDS.
// Stage 1: 3x3 conv as im2col GEMM via v_wmma_f32_16x16x32_f16
//          (M = 2 z-values x 8 pixels, K = 9 taps zero-padded, N = 8 ch).
// Stage 2: per-pixel u_hat (4x4 matmuls) + 3-round dynamic routing with
//          wave32 shfl_xor butterflies (half-wave = pos/app z-split).
// Stage 3: coalesced float4 store of v tile.

typedef __attribute__((ext_vector_type(16))) _Float16 v16h;
typedef __attribute__((ext_vector_type(8)))  float    v8f;

#if defined(__AMDGCN__) && __has_builtin(__builtin_amdgcn_global_load_async_to_lds_b32)
#define ASYNC_LDS 1
typedef __attribute__((address_space(1))) int* as1_i32p;
typedef __attribute__((address_space(3))) int* as3_i32p;
#endif

#define HW      128
#define T0c     4
#define T1c     8
#define Zc      32
#define TILE_W  8

// LDS float offsets
#define MP_OFF  0                   // 512  : normalized W_pos  [c][t][j][k]
#define MA_OFF  512                 // 512  : W_app             [c][t][j][k]
#define XS_OFF  1024                // 3840 : x tile [c][z][3][10]
#define US_OFF  4864                // 8192 : conv out u [c][z][t][px]
#define VS_OFF  1024                // 2048 : v tile (aliases XS after conv)
#define SM_FLOATS 13056             // 52224 bytes

__global__ __launch_bounds__(256) void caps2d_matwo_fused(
    const float* __restrict__ x, const float* __restrict__ Wconv,
    const float* __restrict__ Wpos, const float* __restrict__ Wapp,
    const float* __restrict__ bapp, float* __restrict__ out)
{
  __shared__ float sm[SM_FLOATS];
  const int tid  = threadIdx.x;
  const int wave = tid >> 5;
  const int lane = tid & 31;
  const int n    = blockIdx.z;
  const int hg   = blockIdx.y;
  const int w0   = blockIdx.x * TILE_W;

  // ---- Stage 0a: x halo tile (rows hg-1..hg+1, cols w0-1..w0+8), zero-padded.
  // In-bounds elements go straight to LDS via the async DMA path (ASYNCcnt);
  // padding elements are plain DS zero-stores (disjoint addresses -> no race).
  for (int idx = tid; idx < T0c * Zc * 3 * 10; idx += 256) {
    const int c   = idx / 960;
    const int rem = idx - c * 960;
    const int z   = rem / 30;
    const int r2  = rem - z * 30;
    const int dh  = r2 / 10, dw = r2 - dh * 10;
    const int gh  = hg - 1 + dh, gw = w0 - 1 + dw;
    if (gh >= 0 && gh < HW && gw >= 0 && gw < HW) {
      const size_t gidx =
          ((size_t)((n * T0c + c) * Zc + z)) * (HW * HW) + (size_t)gh * HW + gw;
#ifdef ASYNC_LDS
      __builtin_amdgcn_global_load_async_to_lds_b32(
          (as1_i32p)(x + gidx),
          (as3_i32p)(&sm[XS_OFF + idx]),
          0, 0);
#else
      sm[XS_OFF + idx] = x[gidx];
#endif
    } else {
      sm[XS_OFF + idx] = 0.f;
    }
  }

  // ---- Stage 0b: weights into LDS (mult_pos normalized over j per (c,t,k)) ----
  for (int idx = tid; idx < 1024; idx += 256) {
    if (idx < 512) {
      const int c = idx >> 7, t = (idx >> 4) & 7, k = idx & 3;
      const float raw = Wpos[idx];              // flat == c*128 + t*16 + j*4 + k
      float nrm = 0.f;
      #pragma unroll
      for (int j2 = 0; j2 < 4; ++j2) {
        const float r2 = Wpos[c*128 + t*16 + j2*4 + k];
        nrm += r2 * r2;
      }
      sm[MP_OFF + idx] = raw / sqrtf(fmaxf(nrm, 1e-12f));
    } else {
      sm[MA_OFF + (idx - 512)] = Wapp[idx - 512];   // no normalization
    }
  }

#ifdef ASYNC_LDS
#if __has_builtin(__builtin_amdgcn_s_wait_asynccnt)
  __builtin_amdgcn_s_wait_asynccnt(0);
#else
  asm volatile("s_wait_asynccnt 0x0" ::: "memory");
#endif
#endif
  __syncthreads();

  // ---- Stage 1: conv via WMMA. wave -> (c = wave/2, z-half = wave%2) ----
  {
    const int c    = wave >> 1;
    const int zh   = wave & 1;
    const int half = lane >> 4;        // K-split half of the wave
    const int m    = lane & 15;        // M row owned when building A
    const int px   = m & 7, zs = m >> 3;
    const int col  = lane & 15;        // N column owned when building B

    // B: K x N = 32 x 16. lanes 0-15 hold K=0..15 (halfs 0..15), lanes 16-31 K>=16 (zero).
    v16h B;
    #pragma unroll
    for (int q = 0; q < 16; ++q) B[q] = (_Float16)0.f;
    if (half == 0 && col < T1c) {
      #pragma unroll
      for (int j = 0; j < 9; ++j)
        B[j] = (_Float16)Wconv[(c * 9 + j) * T1c + col];   // W_conv[c][dy][dx][0][t]
    }

    for (int q = 0; q < 8; ++q) {
      const int z0 = zh * 16 + q * 2;                      // 2 z-values per WMMA
      const float* xr = &sm[XS_OFF + c * 960 + (z0 + zs) * 30];
      // A: 16x32 f16. lanes 0-15: halfs 0..8 = K 0..8 (taps 0..8);
      //               lanes 16-31: zero. Rest zero.
      v16h A;
      #pragma unroll
      for (int q2 = 0; q2 < 16; ++q2) A[q2] = (_Float16)0.f;
      if (half == 0) {
        A[0] = (_Float16)xr[0 * 10 + px + 0];
        A[1] = (_Float16)xr[0 * 10 + px + 1];
        A[2] = (_Float16)xr[0 * 10 + px + 2];
        A[3] = (_Float16)xr[1 * 10 + px + 0];
        A[4] = (_Float16)xr[1 * 10 + px + 1];
        A[5] = (_Float16)xr[1 * 10 + px + 2];
        A[6] = (_Float16)xr[2 * 10 + px + 0];
        A[7] = (_Float16)xr[2 * 10 + px + 1];
        A[8] = (_Float16)xr[2 * 10 + px + 2];
      }
      v8f Cz;
      #pragma unroll
      for (int q2 = 0; q2 < 8; ++q2) Cz[q2] = 0.f;
      v8f D = __builtin_amdgcn_wmma_f32_16x16x32_f16(
          false, A, false, B, (short)0, Cz, false, false);
      // D layout: lane l, vgpr v -> row M = v + 8*(l/16), col N = l%16.
      const int t = lane & 15;
      if (t < T1c) {
        float* ur = &sm[US_OFF + ((c * Zc + (z0 + half)) * T1c + t) * TILE_W];
        #pragma unroll
        for (int v2 = 0; v2 < 8; ++v2) ur[v2] = D[v2];     // pixel = v2
      }
    }
  }
  __syncthreads();

  // ---- Stage 2: per-pixel u_hat + routing. wave = pixel, lane = z (0..31) ----
  {
    const int px = wave;
    const int zz = lane;
    const int wg = w0 + px;
    const int zl = zz & 15;
    const int i4 = (zl >> 2) << 2;     // i*4
    const int k  = zl & 3;
    const float cw = (float)wg * (1.f / 128.f);
    const float ch = (float)hg * (1.f / 128.f);

    float uh[32];
    if (zz < 16) {                      // pos block: flat e = t*16 + i*4 + j
      for (int c = 0; c < 4; ++c)
        for (int t = 0; t < 8; ++t) {
          float acc = 0.f;
          #pragma unroll
          for (int j = 0; j < 4; ++j) {
            const int e = t * 16 + i4 + j;
            const float uv = sm[US_OFF + ((c * Zc + (e & 31)) * T1c + (e >> 5)) * TILE_W + px];
            float mv = sm[MP_OFF + (c * 8 + t) * 16 + j * 4 + k];
            if (j == 3) { if (k == 0) mv += cw; else if (k == 1) mv += ch; }
            acc += uv * mv;
          }
          uh[c * 8 + t] = acc;
        }
    } else {                            // app block: flat e = 128 + t*16 + i*4 + j
      for (int c = 0; c < 4; ++c)
        for (int t = 0; t < 8; ++t) {
          const float bac = bapp[c * 8 + t];
          float acc = 0.f;
          #pragma unroll
          for (int j = 0; j < 4; ++j) {
            const int e = 128 + t * 16 + i4 + j;
            const float uv = sm[US_OFF + ((c * Zc + (e & 31)) * T1c + (e >> 5)) * TILE_W + px] + bac;
            acc += uv * sm[MA_OFF + (c * 8 + t) * 16 + j * 4 + k];
          }
          uh[c * 8 + t] = acc;
        }
    }

    float bb[32];
    #pragma unroll
    for (int q = 0; q < 32; ++q) bb[q] = 0.f;
    float vv[8];

    for (int iter = 0; iter < 3; ++iter) {
      float p[8];
      #pragma unroll
      for (int t = 0; t < 8; ++t) {
        float acc = 0.f;
        #pragma unroll
        for (int c = 0; c < 4; ++c) {
          const float r = 1.f / (1.f + expf(-bb[c * 8 + t]));   // sigmoid
          acc += uh[c * 8 + t] * r;
        }
        p[t] = acc;
      }
      // half-wave reductions: lanes 0-15 = pos (max|p|), lanes 16-31 = app (sum p^2)
      #pragma unroll
      for (int t = 0; t < 8; ++t) {
        float mx = fabsf(p[t]);
        float sq = p[t] * p[t];
        #pragma unroll
        for (int mk = 1; mk <= 8; mk <<= 1) {
          mx = fmaxf(mx, __shfl_xor(mx, mk, 32));
          sq += __shfl_xor(sq, mk, 32);
        }
        vv[t] = (zz < 16) ? (p[t] / mx)
                          : (sq / (1.f + sq)) * p[t] * rsqrtf(sq + 1e-9f);
      }
      if (iter < 2) {
        // rout[c,t] = (sum_{z<16} uh*v) * (sum_{z>=16} uh*v); xor-16 swaps halves
        #pragma unroll
        for (int q = 0; q < 32; ++q) {
          float part = uh[q] * vv[q & 7];
          #pragma unroll
          for (int mk = 1; mk <= 8; mk <<= 1)
            part += __shfl_xor(part, mk, 32);
          const float other = __shfl_xor(part, 16, 32);
          bb[q] += part * other;
        }
      } else {
        #pragma unroll
        for (int t = 0; t < 8; ++t)
          sm[VS_OFF + (zz * 8 + t) * TILE_W + px] = vv[t];
      }
    }
  }
  __syncthreads();

  // ---- Stage 3: float4 store of v -> out[n][t][z][h][w] (b128 stores) ----
  for (int q = tid; q < T1c * Zc * 2; q += 256) {
    const int t   = q >> 6;
    const int zz2 = (q >> 1) & 31;
    const int hs  = q & 1;                       // which half of the 8-px row
    const float4 val = *(const float4*)&sm[VS_OFF + (zz2 * 8 + t) * TILE_W + hs * 4];
    *(float4*)&out[((size_t)((n * T1c + t) * Zc + zz2)) * (HW * HW)
                   + (size_t)hg * HW + w0 + hs * 4] = val;
  }
}

extern "C" void kernel_launch(void* const* d_in, const int* in_sizes, int n_in,
                              void* d_out, int out_size, void* d_ws, size_t ws_size,
                              hipStream_t stream) {
  (void)in_sizes; (void)n_in; (void)out_size; (void)d_ws; (void)ws_size;
  const float* x     = (const float*)d_in[0];
  const float* Wconv = (const float*)d_in[1];
  const float* Wpos  = (const float*)d_in[2];
  const float* Wapp  = (const float*)d_in[3];
  const float* bapp  = (const float*)d_in[4];
  float* out = (float*)d_out;

  dim3 grid(HW / TILE_W, HW, 2);   // (w-tiles, h, n) = (16, 128, 2)
  dim3 block(256);                 // 8 wave32 waves
  caps2d_matwo_fused<<<grid, block, 0, stream>>>(x, Wconv, Wpos, Wapp, bapp, out);
}
